// AMRDyGFormer_70325794505142
// MI455X (gfx1250) — matrix operand order repaired and checked by reading, not compiled, not run on hardware
//
#include <hip/hip_runtime.h>
#include <math.h>

// ---------------- problem dims ----------------
constexpr int kT = 16, kN = 6400, kE = 65536, kIN = 128, kED = 16;
constexpr int kH = 256, kB = 64, kHEADS = 8, kDH = 32, kL = 2;
constexpr int kDFF = 2048, kOUT = 8, kS = kT + 1; // 17
constexpr int kKpad = 292; // LDS f32 row stride: mult of 4 (16B align), 292%64=36 -> 16 distinct bank groups

typedef __attribute__((ext_vector_type(16))) _Float16 v16h;
typedef __attribute__((ext_vector_type(8)))  float    v8f;
typedef __attribute__((ext_vector_type(4)))  float    v4f;

// ---------------- small helpers ----------------
__device__ __forceinline__ float blockReduceSum(float v, float* sm) {
  // blockDim.x == 256 (8 waves of 32)
  for (int o = 16; o > 0; o >>= 1) v += __shfl_down(v, o, 32);
  int wid = threadIdx.x >> 5, lid = threadIdx.x & 31;
  if (lid == 0) sm[wid] = v;
  __syncthreads();
  float r = 0.f;
  if (threadIdx.x < 8) r = sm[threadIdx.x];
  if (wid == 0) {
    for (int o = 4; o > 0; o >>= 1) r += __shfl_down(r, o, 32);
    if (lid == 0) sm[0] = r;
  }
  __syncthreads();
  r = sm[0];
  __syncthreads();
  return r;
}

// monotonic float<->uint encoding for atomicMax on floats
__device__ __forceinline__ unsigned fenc(float f) {
  unsigned u = __float_as_uint(f);
  return (u & 0x80000000u) ? ~u : (u | 0x80000000u);
}
__device__ __forceinline__ float fdec(unsigned u) {
  return __uint_as_float((u & 0x80000000u) ? (u & 0x7FFFFFFFu) : ~u);
}

// ---------------- weight pre-pack: f32 [K,Nc] -> f16 WMMA B-fragment layout ----------------
// Wp[((tileN*KC + kc)*32 + lane)*16 + i] = W[kc*32 + (lane>>4)*8 + i + (i>=8 ? 8 : 0)][tileN*16 + (lane&15)]
// (zero-padded when k >= K); each lane's 16 f16 are contiguous (32B, aligned)
__global__ void pack_w_kernel(const float* __restrict__ W, int K, int Nc,
                              _Float16* __restrict__ Wp) {
  int KC = (K + 31) >> 5;
  long idx = (long)blockIdx.x * blockDim.x + threadIdx.x;
  long total = (long)(Nc / 16) * KC * 32;
  if (idx >= total) return;
  int lane = (int)(idx & 31);
  long t2 = idx >> 5;
  int kc = (int)(t2 % KC);
  int tileN = (int)(t2 / KC);
  int half = lane >> 4, l15 = lane & 15;
  int col = tileN * 16 + l15;
  _Float16* dst = Wp + idx * 16;
#pragma unroll
  for (int i = 0; i < 16; ++i) {
    int k = kc * 32 + half * 8 + i + (i >= 8 ? 8 : 0);
    dst[i] = (_Float16)((k < K) ? W[(size_t)k * Nc + col] : 0.f);
  }
}

// ---------------- generic WMMA GEMM with packed f16 weights ----------------
// C[M,Nc] = A[M,K] @ W[K,Nc] (+bias, opt relu); K must be a multiple of 32.
__global__ __launch_bounds__(32)
void gemm_pk_wmma(const float* __restrict__ A, const _Float16* __restrict__ Wp,
                  const float* __restrict__ bias, float* __restrict__ C,
                  int M, int Nc, int K, int relu) {
  const int KC = K >> 5;
  const int tileN = blockIdx.x;          // Nc/16
  const int tileM = blockIdx.y;          // M/16
  const int lane = threadIdx.x;
  const int half = lane >> 4, l15 = lane & 15;
  const int arow = tileM * 16 + l15;
  const int bcol = tileN * 16 + l15;
  v8f acc = {};
  for (int kc = 0; kc < KC; ++kc) {
    v16h a;
    const float* Ar = A + (size_t)arow * K + kc * 32 + half * 8;
#pragma unroll
    for (int i = 0; i < 8; ++i) {
      a[i]     = (_Float16)Ar[i];
      a[8 + i] = (_Float16)Ar[16 + i];
    }
    v16h b = *(const v16h*)(Wp + ((size_t)(tileN * KC + kc) * 32 + lane) * 16);
    acc = __builtin_amdgcn_wmma_f32_16x16x32_f16(false, a, false, b, (short)0, acc,
                                                 false, false);
  }
  float bb = bias ? bias[bcol] : 0.0f;
#pragma unroll
  for (int r = 0; r < 8; ++r) {
    int crow = tileM * 16 + half * 8 + r;
    float v = acc[r] + bb;
    if (relu) v = fmaxf(v, 0.f);
    C[(size_t)crow * Nc + bcol] = v;
  }
}

// ---------------- edge-message GEMM: async-LDS staged gather + WMMA + atomic scatter ----------------
// One block (16 waves) per 16-edge tile. The concat(feat[src], ea) rows are DMA'd into
// LDS once via global_load_async_to_lds_b128 (ASYNCcnt), then each wave computes one
// 16-column tile of msg = concat @ W and scatters msg into agg[dst] with f32 atomics.
__global__ __launch_bounds__(512)
void edge_msg_wmma(const float* __restrict__ feat, int Kf,
                   const float* __restrict__ ea,
                   const int* __restrict__ src, const int* __restrict__ dst,
                   const _Float16* __restrict__ Wp, int KC,
                   float* __restrict__ agg) {
  __shared__ __align__(16) float As[16 * kKpad];
  const int tid = threadIdx.x;
  const int tileE = blockIdx.x;
  const int Ktot = Kf + kED;

  // zero-fill K padding [Ktot, KC*32) (regular ds stores; waited at the barrier)
  const int padn = KC * 32 - Ktot;
  for (int g = tid; g < 16 * padn; g += 512) {
    int row = g / padn, kk = Ktot + g % padn;
    As[row * kKpad + kk] = 0.f;
  }
  // async DMA: 16-byte granules of feat rows (gathered) and edge_attr rows
  {
    unsigned ldsbase = (unsigned)(size_t)As;
    const int q4 = Ktot >> 2;                     // 4-f32 granules per row
    for (int g = tid; g < 16 * q4; g += 512) {
      int row = g / q4, k = (g % q4) * 4;
      const float* gsrc;
      if (k < Kf) gsrc = feat + (size_t)src[tileE * 16 + row] * Kf + k;
      else        gsrc = ea + (size_t)(tileE * 16 + row) * kED + (k - Kf);
      unsigned la = ldsbase + (unsigned)((row * kKpad + k) * 4);
      asm volatile("global_load_async_to_lds_b128 %0, %1, off"
                   :: "v"(la), "v"((unsigned long long)(size_t)gsrc)
                   : "memory");
    }
    asm volatile("s_wait_asynccnt 0" ::: "memory");
  }
  __syncthreads();

  const int wv = tid >> 5;                        // wave id == N tile (kH/16 == 16)
  const int lane = tid & 31;
  const int half = lane >> 4, l15 = lane & 15;
  const float* Arow = As + l15 * kKpad;
  v8f acc = {};
  for (int kc = 0; kc < KC; ++kc) {
    const float* p0 = Arow + kc * 32 + half * 8;
    v4f x0 = *(const v4f*)p0;
    v4f x1 = *(const v4f*)(p0 + 4);
    v4f x2 = *(const v4f*)(p0 + 16);
    v4f x3 = *(const v4f*)(p0 + 20);
    v16h a;
#pragma unroll
    for (int i = 0; i < 4; ++i) {
      a[i]      = (_Float16)x0[i];
      a[4 + i]  = (_Float16)x1[i];
      a[8 + i]  = (_Float16)x2[i];
      a[12 + i] = (_Float16)x3[i];
    }
    v16h b = *(const v16h*)(Wp + ((size_t)(wv * KC + kc) * 32 + lane) * 16);
    acc = __builtin_amdgcn_wmma_f32_16x16x32_f16(false, a, false, b, (short)0, acc,
                                                 false, false);
  }
  const int bcol = wv * 16 + l15;
#pragma unroll
  for (int r = 0; r < 8; ++r) {
    int e = tileE * 16 + half * 8 + r;
    int d = dst[e];
    atomicAdd(&agg[(size_t)d * kH + bcol], acc[r]);
  }
}

// ---------------- misc elementwise kernels ----------------
__global__ void fill_kernel(float* p, float v, long n) {
  long i = (long)blockIdx.x * blockDim.x + threadIdx.x;
  if (i < n) p[i] = v;
}

__global__ void count_kernel(const int* __restrict__ dst, float* __restrict__ cnt) {
  int e = blockIdx.x * blockDim.x + threadIdx.x;
  atomicAdd(&cnt[dst[e]], 1.0f);
}

// out = LayerNorm(relu(agg/max(cnt,1) + (cnt>0 ? neigh_b : 0) + root))
__global__ __launch_bounds__(256)
void combine_relu_ln(const float* __restrict__ agg, const float* __restrict__ cnt,
                     const float* __restrict__ nb, const float* __restrict__ root,
                     const float* __restrict__ g, const float* __restrict__ beta,
                     float* __restrict__ out) {
  __shared__ float sm[8];
  int n = blockIdx.x, t = threadIdx.x;
  float c = cnt[n];
  float v = agg[(size_t)n * kH + t] / fmaxf(c, 1.f) + (c > 0.f ? nb[t] : 0.f)
          + root[(size_t)n * kH + t];
  v = fmaxf(v, 0.f);
  float s = blockReduceSum(v, sm);
  float sq = blockReduceSum(v * v, sm);
  float m = s / kH, var = sq / kH - m * m;
  out[(size_t)n * kH + t] = (v - m) * rsqrtf(var + 1e-5f) * g[t] + beta[t];
}

__global__ __launch_bounds__(256)
void residual_ln(float* __restrict__ seq, const float* __restrict__ add,
                 const float* __restrict__ g, const float* __restrict__ beta) {
  __shared__ float sm[8];
  int row = blockIdx.x, t = threadIdx.x;
  float v = seq[(size_t)row * kH + t] + add[(size_t)row * kH + t];
  float s = blockReduceSum(v, sm);
  float sq = blockReduceSum(v * v, sm);
  float m = s / kH, var = sq / kH - m * m;
  seq[(size_t)row * kH + t] = (v - m) * rsqrtf(var + 1e-5f) * g[t] + beta[t];
}

// s[n] = tanh(pp[n,:]) . score_w + score_b
__global__ __launch_bounds__(32)
void pool_score(const float* __restrict__ pp, const float* __restrict__ sw,
                const float* __restrict__ sb, float* __restrict__ s) {
  int n = blockIdx.x, lane = threadIdx.x;
  float acc = 0.f;
  for (int i = lane; i < kH; i += 32) acc += tanhf(pp[(size_t)n * kH + i]) * sw[i];
  for (int o = 16; o > 0; o >>= 1) acc += __shfl_down(acc, o, 32);
  if (lane == 0) s[n] = acc + sb[0];
}

__global__ void segmax_kernel(const float* __restrict__ s, const int* __restrict__ bt,
                              unsigned* __restrict__ smax) {
  int n = blockIdx.x * blockDim.x + threadIdx.x;
  atomicMax(&smax[bt[n]], fenc(s[n]));
}

__global__ void segexp_kernel(const float* __restrict__ s, const int* __restrict__ bt,
                              const unsigned* __restrict__ smax, float* __restrict__ e,
                              float* __restrict__ segsum) {
  int n = blockIdx.x * blockDim.x + threadIdx.x;
  int b = bt[n];
  float ev = expf(s[n] - fdec(smax[b]));
  e[n] = ev;
  atomicAdd(&segsum[b], ev);
}

__global__ __launch_bounds__(256)
void wpool_kernel(const float* __restrict__ h1, const float* __restrict__ e,
                  const float* __restrict__ segsum, const int* __restrict__ bt,
                  float* __restrict__ pooled_t) {
  int n = blockIdx.x, t = threadIdx.x;
  int b = bt[n];
  float w = e[n] / segsum[b];
  atomicAdd(&pooled_t[(size_t)b * kH + t], h1[(size_t)n * kH + t] * w);
}

__global__ __launch_bounds__(256)
void build_seq(const float* __restrict__ pooled, const float* __restrict__ cls,
               const float* __restrict__ pos, float* __restrict__ seq) {
  int b = blockIdx.x, si = blockIdx.y, t = threadIdx.x;
  float v = (si == 0) ? cls[t] : pooled[((size_t)(si - 1) * kB + b) * kH + t];
  seq[((size_t)b * kS + si) * kH + t] = v + pos[(size_t)si * kH + t];
}

// per (batch, head) small attention: S=17, DH=32
__global__ __launch_bounds__(32)
void attn_kernel(const float* __restrict__ qkv, float* __restrict__ o) {
  int b = blockIdx.x, hd = blockIdx.y, lane = threadIdx.x;
  __shared__ float qs[kS][kDH], ks[kS][kDH], vs[kS][kDH];
  for (int idx = lane; idx < kS * kDH; idx += 32) {
    int si = idx >> 5, di = idx & 31;
    const float* row = qkv + (size_t)(b * kS + si) * (3 * kH);
    qs[si][di] = row[hd * kDH + di];
    ks[si][di] = row[kH + hd * kDH + di];
    vs[si][di] = row[2 * kH + hd * kDH + di];
  }
  __syncthreads();
  if (lane < kS) {
    float sc[kS];
    float mx = -1e30f;
    const float scale = 0.17677669529663687f; // 1/sqrt(32)
    for (int j = 0; j < kS; ++j) {
      float d = 0.f;
      for (int kk = 0; kk < kDH; ++kk) d += qs[lane][kk] * ks[j][kk];
      sc[j] = d * scale;
      mx = fmaxf(mx, sc[j]);
    }
    float se = 0.f;
    for (int j = 0; j < kS; ++j) { sc[j] = expf(sc[j] - mx); se += sc[j]; }
    float inv = 1.f / se;
    for (int kk = 0; kk < kDH; ++kk) {
      float acc = 0.f;
      for (int j = 0; j < kS; ++j) acc += sc[j] * vs[j][kk];
      o[(size_t)(b * kS + lane) * kH + hd * kDH + kk] = acc * inv;
    }
  }
}

__global__ __launch_bounds__(32)
void head_softplus(const float* __restrict__ seq, const float* __restrict__ hw,
                   const float* __restrict__ hb, float* __restrict__ out) {
  int b = blockIdx.x, o = threadIdx.x;
  if (o < kOUT) {
    float acc = hb[o];
    for (int h = 0; h < kH; ++h) acc += seq[((size_t)b * kS) * kH + h] * hw[h * kOUT + o];
    out[b * kOUT + o] = (acc > 20.f) ? acc : log1pf(expf(acc));
  }
}

// ---------------- host-side launch helpers ----------------
static inline void launch_fill(float* p, float v, long n, hipStream_t st) {
  fill_kernel<<<(unsigned)((n + 255) / 256), 256, 0, st>>>(p, v, n);
}
static inline void launch_pack(const float* W, int K, int Nc, _Float16* Wp, hipStream_t st) {
  int KC = (K + 31) >> 5;
  long total = (long)(Nc / 16) * KC * 32;
  pack_w_kernel<<<(unsigned)((total + 255) / 256), 256, 0, st>>>(W, K, Nc, Wp);
}
static inline void launch_gemm(const float* A, const _Float16* Wp, const float* bias,
                               float* C, int M, int Nc, int K, int relu, hipStream_t st) {
  gemm_pk_wmma<<<dim3(Nc / 16, M / 16), 32, 0, st>>>(A, Wp, bias, C, M, Nc, K, relu);
}

extern "C" void kernel_launch(void* const* d_in, const int* in_sizes, int n_in,
                              void* d_out, int out_size, void* d_ws, size_t ws_size,
                              hipStream_t stream) {
  const float* x          = (const float*)d_in[0];
  const float* ea         = (const float*)d_in[1];
  const int*   ei         = (const int*)d_in[2];
  const int*   batch      = (const int*)d_in[3];
  const float* neigh_w0   = (const float*)d_in[4];
  const float* neigh_b0   = (const float*)d_in[5];
  const float* root_w0    = (const float*)d_in[6];
  const float* root_b0    = (const float*)d_in[7];
  const float* ln_g0      = (const float*)d_in[8];
  const float* ln_b0      = (const float*)d_in[9];
  const float* neigh_w1   = (const float*)d_in[10];
  const float* neigh_b1   = (const float*)d_in[11];
  const float* root_w1    = (const float*)d_in[12];
  const float* root_b1    = (const float*)d_in[13];
  const float* ln_g1      = (const float*)d_in[14];
  const float* ln_b1      = (const float*)d_in[15];
  const float* pool_proj_w= (const float*)d_in[16];
  const float* pool_proj_b= (const float*)d_in[17];
  const float* pool_sc_w  = (const float*)d_in[18];
  const float* pool_sc_b  = (const float*)d_in[19];
  const float* pos_emb    = (const float*)d_in[20];
  const float* cls_token  = (const float*)d_in[21];
  const float* tx_qkv_w   = (const float*)d_in[22];
  const float* tx_qkv_b   = (const float*)d_in[23];
  const float* tx_out_w   = (const float*)d_in[24];
  const float* tx_out_b   = (const float*)d_in[25];
  const float* tx_ff1_w   = (const float*)d_in[26];
  const float* tx_ff1_b   = (const float*)d_in[27];
  const float* tx_ff2_w   = (const float*)d_in[28];
  const float* tx_ff2_b   = (const float*)d_in[29];
  const float* tx_ln1_g   = (const float*)d_in[30];
  const float* tx_ln1_b   = (const float*)d_in[31];
  const float* tx_ln2_g   = (const float*)d_in[32];
  const float* tx_ln2_b   = (const float*)d_in[33];
  const float* head_w     = (const float*)d_in[34];
  const float* head_b     = (const float*)d_in[35];
  float* out = (float*)d_out;

  // workspace carve-up (floats; every size is a multiple of 8 -> 32B alignment kept)
  float* ws = (float*)d_ws;
  size_t off = 0;
  auto alloc = [&](size_t n) { float* p = ws + off; off += n; return p; };
  auto allocH = [&](size_t n_f16) {
    size_t nfl = ((n_f16 + 15) & ~(size_t)15) / 2;
    _Float16* p = (_Float16*)(ws + off);
    off += nfl;
    return p;
  };
  float* agg    = alloc((size_t)kN * kH);
  float* cnt    = alloc(kN);                 // contiguous with agg: joint zero fill
  float* h0     = alloc((size_t)kN * kH);
  float* h1     = alloc((size_t)kN * kH);
  float* pp     = alloc((size_t)kN * kH);
  float* sarr   = alloc(kN);
  float* earr   = alloc(kN);
  float* smaxf  = alloc(kB);                 // used as unsigned
  float* segsum = alloc(kB);                 // contiguous with smaxf: joint zero fill
  float* pooled = alloc((size_t)kT * kB * kH);
  float* seq    = alloc((size_t)kB * kS * kH);
  float* qkv    = alloc((size_t)kB * kS * 3 * kH);
  float* obuf   = alloc((size_t)kB * kS * kH);
  float* tmpb   = alloc((size_t)kB * kS * kH);
  float* ffb    = alloc((size_t)kB * kS * kDFF);
  unsigned* smax = (unsigned*)smaxf;

  // packed f16 weight fragments: NT * KC * 32 * 16 f16 each
  const int KC_n0 = (kIN + kED + 31) >> 5;   // 5  (K=144 -> 160)
  const int KC_n1 = (kH + kED + 31) >> 5;    // 9  (K=272 -> 288)
  _Float16* neighW0p = allocH((size_t)(kH / 16) * KC_n0 * 512);
  _Float16* rootW0p  = allocH((size_t)(kH / 16) * (kIN / 32) * 512);
  _Float16* neighW1p = allocH((size_t)(kH / 16) * KC_n1 * 512);
  _Float16* rootW1p  = allocH((size_t)(kH / 16) * (kH / 32) * 512);
  _Float16* poolWp   = allocH((size_t)(kH / 16) * (kH / 32) * 512);
  _Float16* qkvWp[kL], *outWp[kL], *ff1Wp[kL], *ff2Wp[kL];
  for (int l = 0; l < kL; ++l) {
    qkvWp[l] = allocH((size_t)(3 * kH / 16) * (kH / 32) * 512);
    outWp[l] = allocH((size_t)(kH / 16) * (kH / 32) * 512);
    ff1Wp[l] = allocH((size_t)(kDFF / 16) * (kH / 32) * 512);
    ff2Wp[l] = allocH((size_t)(kH / 16) * (kDFF / 32) * 512);
  }

  // ---- pack all weights once per launch ----
  launch_pack(neigh_w0, kIN + kED, kH, neighW0p, stream);
  launch_pack(root_w0, kIN, kH, rootW0p, stream);
  launch_pack(neigh_w1, kH + kED, kH, neighW1p, stream);
  launch_pack(root_w1, kH, kH, rootW1p, stream);
  launch_pack(pool_proj_w, kH, kH, poolWp, stream);
  for (int l = 0; l < kL; ++l) {
    launch_pack(tx_qkv_w + (size_t)l * kH * 3 * kH, kH, 3 * kH, qkvWp[l], stream);
    launch_pack(tx_out_w + (size_t)l * kH * kH, kH, kH, outWp[l], stream);
    launch_pack(tx_ff1_w + (size_t)l * kH * kDFF, kH, kDFF, ff1Wp[l], stream);
    launch_pack(tx_ff2_w + (size_t)l * kDFF * kH, kDFF, kH, ff2Wp[l], stream);
  }

  launch_fill(pooled, 0.f, (long)kT * kB * kH, stream);

  for (int t = 0; t < kT; ++t) {
    const float* xt  = x + (size_t)t * kN * kIN;
    const float* eat = ea + (size_t)t * kE * kED;
    const int* src   = ei + (size_t)t * 2 * kE;
    const int* dstp  = src + kE;
    const int* btp   = batch + (size_t)t * kN;

    // ---- EdgeSAGEConv layer 0 ----
    launch_fill(agg, 0.f, (long)kN * kH + kN, stream);
    edge_msg_wmma<<<kE / 16, 512, 0, stream>>>(xt, kIN, eat, src, dstp,
                                               neighW0p, KC_n0, agg);
    count_kernel<<<kE / 256, 256, 0, stream>>>(dstp, cnt);
    launch_gemm(xt, rootW0p, root_b0, h0, kN, kH, kIN, 0, stream);
    combine_relu_ln<<<kN, 256, 0, stream>>>(agg, cnt, neigh_b0, h0, ln_g0, ln_b0, h0);

    // ---- EdgeSAGEConv layer 1 ----
    launch_fill(agg, 0.f, (long)kN * kH + kN, stream);
    edge_msg_wmma<<<kE / 16, 512, 0, stream>>>(h0, kH, eat, src, dstp,
                                               neighW1p, KC_n1, agg);
    count_kernel<<<kE / 256, 256, 0, stream>>>(dstp, cnt);
    launch_gemm(h0, rootW1p, root_b1, h1, kN, kH, kH, 0, stream);
    combine_relu_ln<<<kN, 256, 0, stream>>>(agg, cnt, neigh_b1, h1, ln_g1, ln_b1, h1);

    // ---- attention pooling ----
    launch_gemm(h1, poolWp, pool_proj_b, pp, kN, kH, kH, 0, stream);
    pool_score<<<kN, 32, 0, stream>>>(pp, pool_sc_w, pool_sc_b, sarr);
    launch_fill(smaxf, 0.f, 2 * kB, stream);   // enc(-inf)=0 and segsum=0
    segmax_kernel<<<kN / 256, 256, 0, stream>>>(sarr, btp, smax);
    segexp_kernel<<<kN / 256, 256, 0, stream>>>(sarr, btp, smax, earr, segsum);
    wpool_kernel<<<kN, 256, 0, stream>>>(h1, earr, segsum, btp,
                                         pooled + (size_t)t * kB * kH);
  }

  // ---- temporal transformer ----
  build_seq<<<dim3(kB, kS), 256, 0, stream>>>(pooled, cls_token, pos_emb, seq);
  const int BS = kB * kS; // 1088, multiple of 16
  for (int l = 0; l < kL; ++l) {
    launch_gemm(seq, qkvWp[l], tx_qkv_b + (size_t)l * 3 * kH, qkv, BS, 3 * kH, kH, 0,
                stream);
    attn_kernel<<<dim3(kB, kHEADS), 32, 0, stream>>>(qkv, obuf);
    launch_gemm(obuf, outWp[l], tx_out_b + (size_t)l * kH, tmpb, BS, kH, kH, 0, stream);
    residual_ln<<<BS, 256, 0, stream>>>(seq, tmpb, tx_ln1_g + (size_t)l * kH,
                                        tx_ln1_b + (size_t)l * kH);
    launch_gemm(seq, ff1Wp[l], tx_ff1_b + (size_t)l * kDFF, ffb, BS, kDFF, kH, 1, stream);
    launch_gemm(ffb, ff2Wp[l], tx_ff2_b + (size_t)l * kH, tmpb, BS, kH, kDFF, 0, stream);
    residual_ln<<<BS, 256, 0, stream>>>(seq, tmpb, tx_ln2_g + (size_t)l * kH,
                                        tx_ln2_b + (size_t)l * kH);
  }
  head_softplus<<<kB, 32, 0, stream>>>(seq, head_w, head_b, out);

  (void)in_sizes; (void)n_in; (void)out_size; (void)ws_size;
}
